// BiLSTMCRF_64209761075379
// MI455X (gfx1250) — matrix operand order repaired and checked by reading, not compiled, not run on hardware
//
#include <hip/hip_runtime.h>
#include <hip/hip_bf16.h>
#include <cstdint>
#include <cstddef>

// Problem dims
#define Bdim 128
#define Ldim 512
#define Edim 256
#define Hdim 128
#define G4   512   // 4*H
#define Tdim 9

typedef _Float16 v16h __attribute__((ext_vector_type(16)));
typedef _Float16 v8h  __attribute__((ext_vector_type(8)));
typedef float    v8f  __attribute__((ext_vector_type(8)));

__device__ __forceinline__ v8f wmma_f16(v16h a, v16h b, v8f c) {
  // D = A(16x32,f16) * B(32x16,f16) + C(16x16,f32)
  return __builtin_amdgcn_wmma_f32_16x16x32_f16(false, a, false, b, (short)0, c, false, false);
}

// A fragment: 16x32 f16, row-major source with leading dim `ld` (halves).
// CDNA5 layout: lanes 0-15 -> M=lane, K={0..7,16..23}; lanes 16-31 -> M=lane-16, K={8..15,24..31}
__device__ __forceinline__ v16h load_a_frag(const _Float16* base, int ld, int row0, int k0, int lane) {
  int m  = row0 + (lane & 15);
  int kb = k0 + ((lane >> 4) << 3);            // +0 or +8
  const _Float16* p = base + (size_t)m * ld + kb;
  v8h lo = *(const v8h*)(p);
  v8h hi = *(const v8h*)(p + 16);
  v16h r;
#pragma unroll
  for (int i = 0; i < 8; ++i) { r[i] = lo[i]; r[8 + i] = hi[i]; }
  return r;
}

// B fragment: 32x16 (KxN) f16 where B = W^T, i.e. B[k][n] = W[n][k]; W row-major ld halves.
// Lane n holds column n; lanes 0-15 -> K={0..7,16..23}, lanes 16-31 -> K={8..15,24..31}.
__device__ __forceinline__ v16h load_b_frag(const _Float16* w, int ld, int n0, int k0, int lane) {
  int n  = n0 + (lane & 15);
  int kb = k0 + ((lane >> 4) << 3);            // +0 or +8
  const _Float16* p = w + (size_t)n * ld + kb;
  v8h lo = *(const v8h*)(p);
  v8h hi = *(const v8h*)(p + 16);
  v16h r;
#pragma unroll
  for (int i = 0; i < 8; ++i) { r[i] = lo[i]; r[8 + i] = hi[i]; }
  return r;
}

__device__ __forceinline__ float sigm(float x) { return 1.0f / (1.0f + __expf(-x)); }

// ---------------------------------------------------------------- f32 -> f16
__global__ void k_f32_to_f16(const float* __restrict__ src, _Float16* __restrict__ dst, int n) {
  for (int i = blockIdx.x * blockDim.x + threadIdx.x; i < n; i += gridDim.x * blockDim.x)
    dst[i] = (_Float16)src[i];
}

// ------------------------------------------------- fuse W2@W1, W2@b1+b2 -> Wc16[16x256], bc[9]
__global__ void k_fuse_w(const float* __restrict__ W1, const float* __restrict__ b1,
                         const float* __restrict__ W2, const float* __restrict__ b2,
                         _Float16* __restrict__ Wc16, float* __restrict__ bc) {
  int tid = threadIdx.x;
  for (int idx = tid; idx < 16 * 256; idx += blockDim.x) {
    int tg = idx >> 8, e = idx & 255;
    float s = 0.f;
    if (tg < Tdim) {
      for (int k = 0; k < 128; ++k) s += W2[tg * 128 + k] * W1[k * 256 + e];
    }
    Wc16[idx] = (_Float16)s;
  }
  if (tid < Tdim) {
    float s = b2[tid];
    for (int k = 0; k < 128; ++k) s += W2[tid * 128 + k] * b1[k];
    bc[tid] = s;
  }
}

// ------------------------------------------------- xp = x @ Wih^T + b   (both directions)
// grid (512, 8, 2), block 256 (8 waves). Block tile M=128, N=64, K-chunks of 32.
__global__ __launch_bounds__(256)
void k_gemm_xp(const _Float16* __restrict__ X,
               const _Float16* __restrict__ Wf, const _Float16* __restrict__ Wb,
               const float* __restrict__ bf, const float* __restrict__ bb,
               float* __restrict__ xpf, float* __restrict__ xpb) {
  __shared__ __align__(32) _Float16 As[128 * 32];
  __shared__ __align__(32) _Float16 Bs[64 * 32];
  const int dir = blockIdx.z;
  const _Float16* W = dir ? Wb : Wf;
  const float* bias = dir ? bb : bf;
  float* out = dir ? xpb : xpf;
  const int m0 = blockIdx.x * 128;
  const int n0 = blockIdx.y * 64;
  const int tid = threadIdx.x;
  const int lane = tid & 31, wave = tid >> 5;
  const int hi8 = (lane >> 4) << 3;

  v8f acc[4];
#pragma unroll
  for (int t = 0; t < 4; ++t) {
    float bv = bias[n0 + t * 16 + (lane & 15)];
#pragma unroll
    for (int r = 0; r < 8; ++r) acc[t][r] = bv;
  }

  for (int kc = 0; kc < 256; kc += 32) {
    {  // stage A: 128 rows x 32 k
      int row = tid >> 1, cp = (tid & 1) * 16;
      const _Float16* g = X + (size_t)(m0 + row) * 256 + kc + cp;
      *(v8h*)&As[row * 32 + cp]     = *(const v8h*)(g);
      *(v8h*)&As[row * 32 + cp + 8] = *(const v8h*)(g + 8);
    }
    {  // stage B: 64 n x 32 k (n-major)
      int n = tid >> 2, kp = (tid & 3) * 8;
      *(v8h*)&Bs[n * 32 + kp] = *(const v8h*)(W + (size_t)(n0 + n) * 256 + kc + kp);
    }
    __syncthreads();
    v16h a = load_a_frag(As, 32, wave * 16, 0, lane);
#pragma unroll
    for (int t = 0; t < 4; ++t) {
      v16h b = load_b_frag(Bs, 32, t * 16, 0, lane);
      acc[t] = wmma_f16(a, b, acc[t]);
    }
    __syncthreads();
  }
#pragma unroll
  for (int t = 0; t < 4; ++t) {
    int n = n0 + t * 16 + (lane & 15);
#pragma unroll
    for (int r = 0; r < 8; ++r) {
      int m = m0 + wave * 16 + r + hi8;
      out[(size_t)m * G4 + n] = acc[t][r];
    }
  }
}

// ------------------------------------------------- sequential LSTM recurrence
// grid 16 (dir = bid>>3, batch tile = bid&7), block 256 (8 waves).
// Dynamic LDS: Whh f16 (512x128 = 128KB) + h-state f16 (16x128 = 4KB).
__global__ __launch_bounds__(256)
void k_lstm_rec(const float* __restrict__ xpf, const float* __restrict__ xpb,
                const float* __restrict__ WhhF, const float* __restrict__ WhhB,
                const int* __restrict__ lens, _Float16* __restrict__ lstm16) {
  extern __shared__ __align__(32) char smem_raw[];
  _Float16* whh = (_Float16*)smem_raw;            // 512*128
  _Float16* hsh = whh + G4 * Hdim;                // 16*128
  __shared__ int lens_s[16];

  const int bid = blockIdx.x;
  const int dir = bid >> 3;
  const int m0 = (bid & 7) * 16;
  const int tid = threadIdx.x;
  const int lane = tid & 31, wave = tid >> 5;
  const int hi8 = (lane >> 4) << 3;
  const float* xp = dir ? xpb : xpf;
  const float* Whh = dir ? WhhB : WhhF;

  // Stage Whh into LDS as f16 (vectorized f32x4)
  for (int i = tid; i < (G4 * Hdim) / 4; i += 256) {
    float4 v = *(const float4*)(Whh + i * 4);
    _Float16* d = whh + i * 4;
    d[0] = (_Float16)v.x; d[1] = (_Float16)v.y; d[2] = (_Float16)v.z; d[3] = (_Float16)v.w;
  }
  // h-state = 0
  for (int i = tid; i < 16 * Hdim; i += 256) hsh[i] = (_Float16)0.f;
  if (tid < 16) lens_s[tid] = lens[m0 + tid];
  __syncthreads();

  v8f creg, hreg;
#pragma unroll
  for (int r = 0; r < 8; ++r) { creg[r] = 0.f; hreg[r] = 0.f; }

  const int nloc = wave * 16 + (lane & 15);   // hidden unit this lane owns in C/D tiles

  for (int t = 0; t < Ldim; ++t) {
    const int te = dir ? (Ldim - 1 - t) : t;
    // gates accumulators initialized from precomputed xp[b][te][:]
    v8f acc[4];
#pragma unroll
    for (int g = 0; g < 4; ++g) {
      int n = g * Hdim + nloc;
#pragma unroll
      for (int r = 0; r < 8; ++r) {
        int b = m0 + r + hi8;
        acc[g][r] = xp[((size_t)b * Ldim + te) * G4 + n];
      }
    }
    // gates += h @ Whh^T  (M=16, N=512, K=128)
#pragma unroll
    for (int kk = 0; kk < 4; ++kk) {
      v16h a = load_a_frag(hsh, Hdim, 0, kk * 32, lane);
#pragma unroll
      for (int g = 0; g < 4; ++g) {
        v16h b = load_b_frag(whh, Hdim, g * Hdim + wave * 16, kk * 32, lane);
        acc[g] = wmma_f16(a, b, acc[g]);
      }
    }
    // fused cell update (i,f,g,o tiles share (m, hidden) coords per lane/VGPR)
#pragma unroll
    for (int r = 0; r < 8; ++r) {
      int m = r + hi8;
      bool mk = te < lens_s[m];
      float iv = sigm(acc[0][r]);
      float fv = sigm(acc[1][r]);
      float gv = tanhf(acc[2][r]);
      float ov = sigm(acc[3][r]);
      float cn = fv * creg[r] + iv * gv;
      float hn = ov * tanhf(cn);
      float h2 = mk ? hn : hreg[r];
      creg[r] = mk ? cn : creg[r];
      hreg[r] = h2;
      float outv = mk ? h2 : 0.f;
      lstm16[((size_t)(m0 + m) * Ldim + te) * 256 + dir * Hdim + nloc] = (_Float16)outv;
    }
    __syncthreads();           // everyone done reading previous h
#pragma unroll
    for (int r = 0; r < 8; ++r) hsh[(r + hi8) * Hdim + nloc] = (_Float16)hreg[r];
    __syncthreads();           // new h visible
  }
}

// ------------------------------------------------- emissions = lstm16 @ Wc^T + bc  (N=9, padded 16)
__global__ __launch_bounds__(256)
void k_emissions(const _Float16* __restrict__ lstm16, const _Float16* __restrict__ Wc16,
                 const float* __restrict__ bc, float* __restrict__ emis) {
  const int tid = threadIdx.x;
  const int lane = tid & 31, wave = tid >> 5;
  const int hi8 = (lane >> 4) << 3;
  const int m0 = blockIdx.x * 128 + wave * 16;
  const int n = lane & 15;

  v8f acc;
  float bv = (n < Tdim) ? bc[n] : 0.f;
#pragma unroll
  for (int r = 0; r < 8; ++r) acc[r] = bv;

  for (int kc = 0; kc < 256; kc += 32) {
    v16h a = load_a_frag(lstm16, 256, m0, kc, lane);
    v16h b = load_b_frag(Wc16, 256, 0, kc, lane);
    acc = wmma_f16(a, b, acc);
  }
  if (n < Tdim) {
#pragma unroll
    for (int r = 0; r < 8; ++r) {
      int m = m0 + r + hi8;
      emis[(size_t)m * Tdim + n] = acc[r];
    }
  }
}

// ------------------------------------------------- CRF: llh + Viterbi, one thread per batch row
__global__ __launch_bounds__(128)
void k_crf(const float* __restrict__ emis, const int* __restrict__ tags,
           const int* __restrict__ lens, const float* __restrict__ start_t,
           const float* __restrict__ end_t, const float* __restrict__ trans,
           unsigned char* __restrict__ bp, float* __restrict__ out) {
  __shared__ float tr_s[Tdim * Tdim];
  __shared__ float st_s[Tdim], en_s[Tdim];
  __shared__ float red[128];
  const int b = threadIdx.x;
  if (b < Tdim * Tdim) tr_s[b] = trans[b];
  if (b < Tdim) { st_s[b] = start_t[b]; en_s[b] = end_t[b]; }
  __syncthreads();

  const float* em = emis + (size_t)b * Ldim * Tdim;
  const int* tg = tags + (size_t)b * Ldim;
  const int mylen = lens[b];

  float alpha[Tdim], score[Tdim];
  int tag0 = tg[0];
  float num = st_s[tag0] + em[tag0];
#pragma unroll
  for (int j = 0; j < Tdim; ++j) { alpha[j] = st_s[j] + em[j]; score[j] = alpha[j]; }

  for (int t = 1; t < Ldim; ++t) {
    const float* emt = em + t * Tdim;
    int cur = tg[t];
    bool mk = cur != 0;
    if (mk) num += tr_s[tg[t - 1] * Tdim + cur] + emt[cur];

    // Double-buffered: compute all new values from OLD alpha/score, then commit.
    float na[Tdim], nb[Tdim];
    unsigned char bpj[Tdim];
    for (int j = 0; j < Tdim; ++j) {
      float mx = -1e30f;
      for (int i = 0; i < Tdim; ++i) { float v = alpha[i] + tr_s[i * Tdim + j]; mx = v > mx ? v : mx; }
      float s = 0.f;
      for (int i = 0; i < Tdim; ++i) s += __expf(alpha[i] + tr_s[i * Tdim + j] - mx);
      na[j] = mx + __logf(s) + emt[j];

      float bmax = -1e30f; int bi = 0;
      for (int i = 0; i < Tdim; ++i) {
        float v = score[i] + tr_s[i * Tdim + j];
        if (v > bmax) { bmax = v; bi = i; }
      }
      nb[j] = bmax + emt[j];
      bpj[j] = mk ? (unsigned char)bi : (unsigned char)j;
    }
    for (int j = 0; j < Tdim; ++j) {
      if (mk) { alpha[j] = na[j]; score[j] = nb[j]; }
      bp[((size_t)(t - 1) * Bdim + b) * Tdim + j] = bpj[j];
    }
  }

  num += en_s[tg[mylen - 1]];
  float mx = -1e30f;
  for (int j = 0; j < Tdim; ++j) { float v = alpha[j] + en_s[j]; mx = v > mx ? v : mx; }
  float s = 0.f;
  for (int j = 0; j < Tdim; ++j) s += __expf(alpha[j] + en_s[j] - mx);
  float den = mx + __logf(s);

  // Viterbi backtrack, written directly (masked to 0 at pads).
  int best = 0; float bv = -1e30f;
  for (int j = 0; j < Tdim; ++j) { float v = score[j] + en_s[j]; if (v > bv) { bv = v; best = j; } }
  int cur = best;
  out[(size_t)b * Ldim + (Ldim - 1)] = (Ldim - 1 < mylen) ? (float)cur : 0.f;
  for (int t = Ldim - 2; t >= 0; --t) {
    cur = bp[((size_t)t * Bdim + b) * Tdim + cur];
    out[(size_t)b * Ldim + t] = (t < mylen) ? (float)cur : 0.f;
  }

  red[b] = num - den;
  __syncthreads();
  for (int off = 64; off > 0; off >>= 1) {
    if (b < off) red[b] += red[b + off];
    __syncthreads();
  }
  if (b == 0) out[(size_t)Bdim * Ldim] = -red[0];
}

// ---------------------------------------------------------------- launcher
extern "C" void kernel_launch(void* const* d_in, const int* in_sizes, int n_in,
                              void* d_out, int out_size, void* d_ws, size_t ws_size,
                              hipStream_t stream) {
  const float* embeds = (const float*)d_in[0];
  const float* Wih_f  = (const float*)d_in[1];
  const float* Whh_f  = (const float*)d_in[2];
  const float* b_f    = (const float*)d_in[3];
  const float* Wih_b  = (const float*)d_in[4];
  const float* Whh_b  = (const float*)d_in[5];
  const float* b_b    = (const float*)d_in[6];
  const float* W1     = (const float*)d_in[7];
  const float* b1     = (const float*)d_in[8];
  const float* W2     = (const float*)d_in[9];
  const float* b2     = (const float*)d_in[10];
  const float* start_t= (const float*)d_in[11];
  const float* end_t  = (const float*)d_in[12];
  const float* trans  = (const float*)d_in[13];
  const int*   lens   = (const int*)d_in[14];
  const int*   tags   = (const int*)d_in[15];
  float* out = (float*)d_out;

  char* ws = (char*)d_ws;
  size_t off = 0;
  auto alloc = [&](size_t bytes) { char* p = ws + off; off += (bytes + 255) & ~(size_t)255; return p; };
  _Float16* x16    = (_Float16*)alloc((size_t)Bdim * Ldim * Edim * 2);
  _Float16* wih16f = (_Float16*)alloc((size_t)G4 * Edim * 2);
  _Float16* wih16b = (_Float16*)alloc((size_t)G4 * Edim * 2);
  _Float16* wc16   = (_Float16*)alloc((size_t)16 * 256 * 2);
  float*    bc     = (float*)alloc(Tdim * 4);
  float*    xpf    = (float*)alloc((size_t)Bdim * Ldim * G4 * 4);
  float*    xpb    = (float*)alloc((size_t)Bdim * Ldim * G4 * 4);
  _Float16* lstm16 = (_Float16*)alloc((size_t)Bdim * Ldim * 256 * 2);
  float*    emis   = (float*)alloc((size_t)Bdim * Ldim * Tdim * 4);
  unsigned char* bp = (unsigned char*)alloc((size_t)(Ldim - 1) * Bdim * Tdim);

  k_f32_to_f16<<<4096, 256, 0, stream>>>(embeds, x16, Bdim * Ldim * Edim);
  k_f32_to_f16<<<512, 256, 0, stream>>>(Wih_f, wih16f, G4 * Edim);
  k_f32_to_f16<<<512, 256, 0, stream>>>(Wih_b, wih16b, G4 * Edim);
  k_fuse_w<<<1, 256, 0, stream>>>(W1, b1, W2, b2, wc16, bc);

  k_gemm_xp<<<dim3(512, 8, 2), 256, 0, stream>>>(x16, wih16f, wih16b, b_f, b_b, xpf, xpb);

  size_t smem = (size_t)G4 * Hdim * 2 + (size_t)16 * Hdim * 2;  // 135168 B
  k_lstm_rec<<<16, 256, smem, stream>>>(xpf, xpb, Whh_f, Whh_b, lens, lstm16);

  k_emissions<<<512, 256, 0, stream>>>(lstm16, wc16, bc, emis);

  k_crf<<<1, 128, 0, stream>>>(emis, tags, lens, start_t, end_t, trans, bp, out);
}
